// PointNetSetAbstraction_84043920048191
// MI455X (gfx1250) — matrix operand order repaired and checked by reading, not compiled, not run on hardware
//
#include <hip/hip_runtime.h>
#include <hip/hip_fp16.h>

typedef __attribute__((ext_vector_type(16))) _Float16 v16h;
typedef __attribute__((ext_vector_type(8)))  _Float16 v8h;
typedef __attribute__((ext_vector_type(8)))  float    v8f;

#define N_PTS  16384
#define S_CENT 4096
#define KNBR   32
#define RAD2   0.04f

// ======================= weight prep: f32 -> f16 B-fragment layout =======================
// B fragment (16x16x32 f16): frag (kt,nt); lane l holds col n = nt*16 + (l&15),
// rows k = kt*32 + (i<8 ? (l>>4)*8 + i : 16 + (l>>4)*8 + (i-8)) for i=0..15, stored contiguously [frag][lane][i].
__global__ void prep_weights_kernel(const float* __restrict__ W1,
                                    const float* __restrict__ W2,
                                    const float* __restrict__ W3,
                                    _Float16* __restrict__ W1B,
                                    _Float16* __restrict__ W2B,
                                    _Float16* __restrict__ W3B) {
  const int total = 12288 + 16384 + 32768;   // (3*8 + 4*8 + 4*16) frags * 512 halfs
  for (int t = blockIdx.x * blockDim.x + threadIdx.x; t < total;
       t += gridDim.x * blockDim.x) {
    const float* W; _Float16* WB; int NT, Krows; int u = t;
    if (u < 12288)      { W = W1; WB = W1B; NT = 8;  Krows = 67;  }
    else if (u < 28672) { u -= 12288; W = W2; WB = W2B; NT = 8;  Krows = 128; }
    else                { u -= 28672; W = W3; WB = W3B; NT = 16; Krows = 128; }
    int i    = u & 15;
    int lane = (u >> 4) & 31;
    int frag = u >> 9;
    int nt = frag % NT, kt = frag / NT;
    int g = lane >> 4;
    int k = kt * 32 + (i < 8 ? g * 8 + i : 16 + g * 8 + (i - 8));
    int n = nt * 16 + (lane & 15);
    int N = NT * 16;
    float v = (k < Krows) ? W[k * N + n] : 0.0f;  // zero-pad K rows 67..95 of W1
    WB[u] = (_Float16)v;
  }
}

// ======================= FPS: single workgroup, pos resident in LDS =======================
extern "C" __global__ __launch_bounds__(1024)
void fps_kernel(const float* __restrict__ pos, float* __restrict__ newpos) {
  extern __shared__ __align__(16) char dynsm[];
  float* px = (float*)dynsm;
  float* py = px + N_PTS;
  float* pz = py + N_PTS;
  float* redv = pz + N_PTS;            // [32]
  int*   redi = (int*)(redv + 32);     // [32]
  int*   bfar = (int*)(redi + 32);     // [1]
  const int tid = threadIdx.x;

  for (int j = tid; j < N_PTS; j += 1024) {
    px[j] = pos[3 * j + 0];
    py[j] = pos[3 * j + 1];
    pz[j] = pos[3 * j + 2];
  }
  float mind[16];
  #pragma unroll
  for (int j = 0; j < 16; ++j) mind[j] = 3.4e38f;
  __syncthreads();

  int far = 0;
  for (int s = 0; s < S_CENT; ++s) {
    if (tid == 0) {
      newpos[3 * s + 0] = px[far];
      newpos[3 * s + 1] = py[far];
      newpos[3 * s + 2] = pz[far];
    }
    const float fx = px[far], fy = py[far], fz = pz[far];
    float bv = -1.0f; int bi = 0x7fffffff;
    #pragma unroll
    for (int j = 0; j < 16; ++j) {
      const int p = tid + j * 1024;
      const float dx = px[p] - fx, dy = py[p] - fy, dz = pz[p] - fz;
      const float d = dx * dx + dy * dy + dz * dz;
      const float m = mind[j] < d ? mind[j] : d;
      mind[j] = m;
      if (m > bv) { bv = m; bi = p; }   // within-thread indices increase -> strict > is stable
    }
    // wave32 argmax reduce (max value, tie -> smaller index)
    #pragma unroll
    for (int off = 16; off > 0; off >>= 1) {
      const float ov = __shfl_xor(bv, off);
      const int   oi = __shfl_xor(bi, off);
      if (ov > bv || (ov == bv && oi < bi)) { bv = ov; bi = oi; }
    }
    if ((tid & 31) == 0) { redv[tid >> 5] = bv; redi[tid >> 5] = bi; }
    __syncthreads();
    if (tid < 32) {
      bv = redv[tid]; bi = redi[tid];
      #pragma unroll
      for (int off = 16; off > 0; off >>= 1) {
        const float ov = __shfl_xor(bv, off);
        const int   oi = __shfl_xor(bi, off);
        if (ov > bv || (ov == bv && oi < bi)) { bv = ov; bi = oi; }
      }
      if (tid == 0) *bfar = bi;
    }
    __syncthreads();
    far = *bfar;
    __syncthreads();
  }
}

// ======================= ball query: one thread per center, pos in LDS =======================
extern "C" __global__ __launch_bounds__(256)
void ballq_kernel(const float* __restrict__ pos, const float* __restrict__ newpos,
                  int* __restrict__ nbr, int* __restrict__ cnt) {
  extern __shared__ __align__(16) char dynsm[];
  float* px = (float*)dynsm;
  float* py = px + N_PTS;
  float* pz = py + N_PTS;
  float* ld = pz + N_PTS;                  // [32][256] d2 list
  int*   li = (int*)(ld + KNBR * 256);     // [32][256] idx list
  const int tid = threadIdx.x;

  for (int j = tid; j < N_PTS; j += 256) {
    px[j] = pos[3 * j + 0];
    py[j] = pos[3 * j + 1];
    pz[j] = pos[3 * j + 2];
  }
  __syncthreads();

  const int c = blockIdx.x * 256 + tid;
  const float cx = newpos[3 * c + 0], cy = newpos[3 * c + 1], cz = newpos[3 * c + 2];
  int n = 0;
  for (int j = 0; j < N_PTS; ++j) {
    const float dx = px[j] - cx, dy = py[j] - cy, dz = pz[j] - cz;
    const float d2 = dx * dx + dy * dy + dz * dz;
    if (d2 <= RAD2) {
      const bool full = (n >= KNBR);
      const float worst = full ? ld[(KNBR - 1) * 256 + tid] : 3.4e38f;
      if (!full || d2 < worst) {           // strict <: on tie existing (lower idx) wins
        const int lim = full ? KNBR : n;
        int p = lim;
        for (int q = 0; q < lim; ++q) {
          if (ld[q * 256 + tid] > d2) { p = q; break; }  // stable insert position
        }
        int last = full ? (KNBR - 1) : n;
        if (last > KNBR - 1) last = KNBR - 1;
        for (int q = last; q > p; --q) {
          ld[q * 256 + tid] = ld[(q - 1) * 256 + tid];
          li[q * 256 + tid] = li[(q - 1) * 256 + tid];
        }
        ld[p * 256 + tid] = d2;
        li[p * 256 + tid] = j;
        if (!full) ++n;
      }
    }
  }
  for (int k = 0; k < KNBR; ++k)
    nbr[c * KNBR + k] = (k < n) ? li[k * 256 + tid] : 0;
  cnt[c] = n;
}

// ======================= WMMA MLP + masked max aggregate =======================
__device__ __forceinline__ v16h mk16(v8h lo, v8h hi) {
  return __builtin_shufflevector(lo, hi, 0,1,2,3,4,5,6,7,8,9,10,11,12,13,14,15);
}
// A fragment (16x32 f16) from LDS [32][128]: lane l -> row m = mt*16+(l&15),
// K halves {g*8+0..7, 16+g*8+0..7} within ktile, g = l>>4  (ISA 7.12.2 layout)
__device__ __forceinline__ v16h load_a_frag(const _Float16* H, int mt, int kt, int lane) {
  const int m = mt * 16 + (lane & 15);
  const int g = lane >> 4;
  const _Float16* p = H + m * 128 + kt * 32 + g * 8;
  return mk16(*(const v8h*)p, *(const v8h*)(p + 16));
}
__device__ __forceinline__ v16h load_b_frag(const _Float16* WB, int kt, int nt, int NT, int lane) {
  const _Float16* p = WB + (((kt * NT) + nt) * 32 + lane) * 16;
  return mk16(*(const v8h*)p, *(const v8h*)(p + 8));
}

template <int KT, int NT>
__device__ __forceinline__ void gemm_relu_store(const _Float16* A, const _Float16* WB,
                                                const float* bias, _Float16* O, int lane) {
  const int g = lane >> 4, nl = lane & 15;
  #pragma unroll
  for (int nt = 0; nt < NT; ++nt) {
    const float bv = bias[nt * 16 + nl];
    #pragma unroll
    for (int mt = 0; mt < 2; ++mt) {
      v8f acc = {bv, bv, bv, bv, bv, bv, bv, bv};
      #pragma unroll
      for (int kt = 0; kt < KT; ++kt) {
        acc = __builtin_amdgcn_wmma_f32_16x16x32_f16(
            false, load_a_frag(A, mt, kt, lane),
            false, load_b_frag(WB, kt, nt, NT, lane),
            (short)0, acc, false, false);
      }
      #pragma unroll
      for (int r = 0; r < 8; ++r) {
        float v = acc[r];
        v = v > 0.f ? v : 0.f;                                  // ReLU
        O[(mt * 16 + g * 8 + r) * 128 + nt * 16 + nl] = (_Float16)v;
      }
    }
  }
}

extern "C" __global__ __launch_bounds__(256)
void mlp_kernel(const float* __restrict__ pos, const float* __restrict__ x,
                const float* __restrict__ newpos,
                const int* __restrict__ nbr, const int* __restrict__ cnt,
                const _Float16* __restrict__ W1B, const _Float16* __restrict__ W2B,
                const _Float16* __restrict__ W3B,
                const float* __restrict__ b1, const float* __restrict__ b2,
                const float* __restrict__ b3,
                float* __restrict__ newx) {
  extern __shared__ __align__(16) char dynsm[];
  _Float16* smem = (_Float16*)dynsm;
  const int w = threadIdx.x >> 5, lane = threadIdx.x & 31;
  _Float16* buf0 = smem + w * 32 * 128;               // 8 waves x [32][128]
  _Float16* buf1 = smem + 8 * 32 * 128 + w * 32 * 128;
  const int c = blockIdx.x * 8 + w;
  const int n_valid = cnt[c];

  // ---- phase 1: gather edge features into H0 (f16): [x_j(64) | rel(3) | 0 pad to 96] ----
  {
    const int j  = lane;                    // neighbor slot
    const int nb = nbr[c * KNBR + j];
    const float4* xr = (const float4*)(x + (long)nb * 64);
    _Float16* h0 = buf0 + j * 128;
    #pragma unroll
    for (int t = 0; t < 16; ++t) {
      const float4 v = xr[t];
      h0[4 * t + 0] = (_Float16)v.x;
      h0[4 * t + 1] = (_Float16)v.y;
      h0[4 * t + 2] = (_Float16)v.z;
      h0[4 * t + 3] = (_Float16)v.w;
    }
    h0[64] = (_Float16)(pos[nb * 3 + 0] - newpos[c * 3 + 0]);
    h0[65] = (_Float16)(pos[nb * 3 + 1] - newpos[c * 3 + 1]);
    h0[66] = (_Float16)(pos[nb * 3 + 2] - newpos[c * 3 + 2]);
    #pragma unroll
    for (int t = 67; t < 96; ++t) h0[t] = (_Float16)0.f;
  }
  __syncthreads();

  gemm_relu_store<3, 8>(buf0, W1B, b1, buf1, lane);   // 67(->96) -> 128
  __syncthreads();
  gemm_relu_store<4, 8>(buf1, W2B, b2, buf0, lane);   // 128 -> 128
  __syncthreads();

  // ---- layer 3 + masked max over neighbor slots (valid slots are a prefix) ----
  {
    const int g = lane >> 4, nl = lane & 15;
    #pragma unroll
    for (int nt = 0; nt < 16; ++nt) {
      const float bv = b3[nt * 16 + nl];
      float cm = -1.0f;
      #pragma unroll
      for (int mt = 0; mt < 2; ++mt) {
        v8f acc = {bv, bv, bv, bv, bv, bv, bv, bv};
        #pragma unroll
        for (int kt = 0; kt < 4; ++kt) {
          acc = __builtin_amdgcn_wmma_f32_16x16x32_f16(
              false, load_a_frag(buf0, mt, kt, lane),
              false, load_b_frag(W3B, kt, nt, 16, lane),
              (short)0, acc, false, false);
        }
        float tm = -1.0f;
        #pragma unroll
        for (int r = 0; r < 8; ++r) {
          const int m = mt * 16 + g * 8 + r;          // neighbor slot of this C element
          float v = acc[r];
          v = v > 0.f ? v : 0.f;                      // ReLU (valid rows are >= 0)
          v = (m < n_valid) ? v : -1.0f;              // mask invalid slots
          tm = tm > v ? tm : v;
        }
        const float ot = __shfl_xor(tm, 16);          // combine row halves (lanes l <-> l+16)
        tm = tm > ot ? tm : ot;
        cm = cm > tm ? cm : tm;
      }
      if (lane < 16) {
        const float o = (n_valid > 0) ? (cm > 0.f ? cm : 0.f) : 0.0f;
        newx[(long)c * 256 + nt * 16 + lane] = o;
      }
    }
  }
}

// ======================= host launch =======================
extern "C" void kernel_launch(void* const* d_in, const int* in_sizes, int n_in,
                              void* d_out, int out_size, void* d_ws, size_t ws_size,
                              hipStream_t stream) {
  const float* pos = (const float*)d_in[0];   // [16384,3]
  const float* x   = (const float*)d_in[1];   // [16384,64]
  const float* W1  = (const float*)d_in[2];   // [67,128]
  const float* b1  = (const float*)d_in[3];   // [128]
  const float* W2  = (const float*)d_in[4];   // [128,128]
  const float* b2  = (const float*)d_in[5];   // [128]
  const float* W3  = (const float*)d_in[6];   // [128,256]
  const float* b3  = (const float*)d_in[7];   // [256]

  float* out    = (float*)d_out;
  float* newpos = out;                        // [4096,3]
  float* newx   = out + S_CENT * 3;           // [4096,256]

  char* ws = (char*)d_ws;
  int*      cntb = (int*)ws;                          // 4096 ints
  int*      nbrb = (int*)(ws + 16384);                // 4096*32 ints
  _Float16* W1B  = (_Float16*)(ws + 16384 + 524288);  // 12288 halfs
  _Float16* W2B  = W1B + 12288;                       // 16384 halfs
  _Float16* W3B  = W2B + 16384;                       // 32768 halfs

  const size_t fps_lds  = (size_t)N_PTS * 3 * 4 + 32 * 4 + 32 * 4 + 16;
  const size_t ballq_lds = (size_t)N_PTS * 3 * 4 + (size_t)KNBR * 256 * 4 * 2;
  const size_t mlp_lds  = (size_t)2 * 8 * 32 * 128 * sizeof(_Float16);

  (void)hipFuncSetAttribute((const void*)fps_kernel,
                            hipFuncAttributeMaxDynamicSharedMemorySize, (int)fps_lds);
  (void)hipFuncSetAttribute((const void*)ballq_kernel,
                            hipFuncAttributeMaxDynamicSharedMemorySize, (int)ballq_lds);
  (void)hipFuncSetAttribute((const void*)mlp_kernel,
                            hipFuncAttributeMaxDynamicSharedMemorySize, (int)mlp_lds);

  prep_weights_kernel<<<64, 256, 0, stream>>>(W1, W2, W3, W1B, W2B, W3B);
  fps_kernel<<<1, 1024, fps_lds, stream>>>(pos, newpos);
  ballq_kernel<<<S_CENT / 256, 256, ballq_lds, stream>>>(pos, newpos, nbrb, cntb);
  mlp_kernel<<<S_CENT / 8, 256, mlp_lds, stream>>>(pos, x, newpos, nbrb, cntb,
                                                   W1B, W2B, W3B, b1, b2, b3, newx);
}
